// Attention_2018634629373
// MI455X (gfx1250) — matrix operand order repaired and checked by reading, not compiled, not run on hardware
//
#include <hip/hip_runtime.h>
#include <hip/hip_bf16.h>
#include <math.h>

// ---------------------------------------------------------------------------
// Attention (Bahdanau-style) for MI455X / gfx1250, wave32 + WMMA bf16.
//   B=16, S=2048, H=1024. Dominant op: enc[32768 x 2048] x We^T[2048 x 2048]
//   consumed on the fly into scores, then masked softmax + weighted sum.
// ---------------------------------------------------------------------------

typedef __bf16 bf16_t;
typedef __attribute__((ext_vector_type(4)))  bf16_t v4bf;
typedef __attribute__((ext_vector_type(8)))  bf16_t v8bf;
typedef __attribute__((ext_vector_type(16))) bf16_t v16bf;
typedef __attribute__((ext_vector_type(8)))  float  v8f;

#define NB      16
#define SEQ     2048
#define HID     1024
#define TWO_H   2048
#define THREE_H 3072

// workspace layout (bytes)
#define WS_WE      0u                 // bf16 [2048][2048]  = 8,388,608 B
#define WS_HPB     8388608u           // f32  [16][2048]    (h_part + b_attn)
#define WS_SCORES  8519680u           // f32  [16][2048]
#define WS_ENERGY  8650752u           // f32  [16][2048]
// total ~8.78 MB of d_ws required

// Branchless tanh: 1 - 2/(exp(2x)+1); exact at saturation, no divergence.
// v_mul + v_exp_f32 + v_add + v_rcp_f32 + v_fma  (vs branchy OCML tanhf)
static __device__ __forceinline__ float tanh_fast(float x) {
  const float e = __builtin_amdgcn_exp2f(x * 2.88539008177793f); // exp(2x)
  return 1.0f - 2.0f * __builtin_amdgcn_rcpf(e + 1.0f);
}

// ---------------------------------------------------------------------------
// Prep 1: We_bf16[o][k] = bf16( W_attn[o][H + k] )      (8 MB, L2-resident)
// ---------------------------------------------------------------------------
__global__ __launch_bounds__(256) void k_convert_we(const float* __restrict__ W,
                                                    bf16_t* __restrict__ We) {
  const int total4 = (TWO_H * TWO_H) / 4;
  for (int idx = blockIdx.x * blockDim.x + threadIdx.x; idx < total4;
       idx += gridDim.x * blockDim.x) {
    const int o  = idx >> 9;          // 512 float4 per row
    const int k4 = idx & 511;
    const float4 v = *(const float4*)(W + (size_t)o * THREE_H + HID + k4 * 4);
    v4bf p;
    p[0] = (bf16_t)v.x; p[1] = (bf16_t)v.y; p[2] = (bf16_t)v.z; p[3] = (bf16_t)v.w;
    *(v4bf*)(We + (size_t)idx * 4) = p;   // idx*4 == o*2048 + k4*4
  }
}

// ---------------------------------------------------------------------------
// Prep 2: hpb[b][o] = b_attn[o] + sum_k W_attn[o][k] * hidden[k][b]
//   grid 128, block (16 b, 16 o)
// ---------------------------------------------------------------------------
__global__ __launch_bounds__(256) void k_hpb(const float* __restrict__ W,
                                             const float* __restrict__ hidden,
                                             const float* __restrict__ b_attn,
                                             float* __restrict__ hpb) {
  const int b = threadIdx.x;
  const int o = blockIdx.x * 16 + threadIdx.y;
  const float* wr = W + (size_t)o * THREE_H;
  float acc = 0.f;
#pragma unroll 4
  for (int k = 0; k < HID; ++k)
    acc += wr[k] * hidden[k * NB + b];
  hpb[b * TWO_H + o] = acc + b_attn[o];
}

// ---------------------------------------------------------------------------
// Zero the output (harness poisons it; context kernel uses atomics).
// ---------------------------------------------------------------------------
__global__ __launch_bounds__(256) void k_zero_out(float* __restrict__ out) {
  const int i = blockIdx.x * 256 + threadIdx.x;
  if (i < NB * TWO_H) out[i] = 0.f;
}

// ---------------------------------------------------------------------------
// Main kernel: scores[m] = sum_o w_v[o] * tanh( enc[m,:]·We[o,:] + hpb[b][o] )
//   512 blocks x 256 threads. Block = 64 rows of one batch, LDS-resident bf16.
//   Wave w handles N-tiles {w, w+8, ...}; one B fragment feeds 4 M-tiles.
// ---------------------------------------------------------------------------
#define M_BLK  64
#define LDS_LD 2056   // 2048 + 8 bf16 pad -> row stride 4112 B, conflict-free

__global__ __launch_bounds__(256) void k_scores(const float* __restrict__ enc,
                                                const bf16_t* __restrict__ We,
                                                const float* __restrict__ hpb,
                                                const float* __restrict__ wv,
                                                float* __restrict__ scores) {
  __shared__ __align__(16) bf16_t encLds[M_BLK * LDS_LD];   // 263,168 B
  __shared__ float scoreLds[M_BLK];

  const int tid  = threadIdx.x;
  const int m0   = blockIdx.x * M_BLK;
  const int bIdx = m0 >> 11;                 // m0 / SEQ (64 | 2048)

  if (tid < M_BLK) scoreLds[tid] = 0.f;

  // ---- stage enc rows [m0, m0+64) as bf16 into LDS --------------------------
  const float* encBase = enc + (size_t)m0 * TWO_H;
  for (int t = tid; t < (M_BLK * TWO_H) / 4; t += 256) {
    const int row = t >> 9;                  // 512 float4 per row
    const int c4  = t & 511;
    const float4 v = *(const float4*)(encBase + (size_t)row * TWO_H + c4 * 4);
    v4bf p;
    p[0] = (bf16_t)v.x; p[1] = (bf16_t)v.y; p[2] = (bf16_t)v.z; p[3] = (bf16_t)v.w;
    *(v4bf*)(&encLds[row * LDS_LD + c4 * 4]) = p;
  }
  __syncthreads();

  const int lane  = tid & 31;
  const int wave  = tid >> 5;
  const int nlane = lane & 15;   // N (and A-row) index within tile
  const int khalf = lane >> 4;   // which K half this lane holds

  // A fragment base per M-subtile (ISA 7.12.2 16-bit A layout)
  const bf16_t* aB0 = &encLds[nlane * LDS_LD + khalf * 8];
  const float* hpb_b = hpb + bIdx * TWO_H;

  float sAcc[4][8];
#pragma unroll
  for (int mt = 0; mt < 4; ++mt)
#pragma unroll
    for (int r = 0; r < 8; ++r) sAcc[mt][r] = 0.f;

#define LOAD_A(p)                                                             \
  ({ v8bf lo_ = *(const v8bf*)(p);                                            \
     v8bf hi_ = *(const v8bf*)((p) + 16);                                     \
     __builtin_shufflevector(lo_, hi_, 0,1,2,3,4,5,6,7,8,9,10,11,12,13,14,15); })

  for (int nt = wave; nt < 128; nt += 8) {
    const int o0 = nt << 4;
    // B fragment: lane = column o0+nlane, k = 8*khalf+{0..7} and 16+8*khalf+{0..7} (matches A)
    const bf16_t* pb = We + (size_t)(o0 + nlane) * TWO_H + khalf * 8;
    if (nt + 8 < 128)
      __builtin_prefetch(pb + (size_t)128 * TWO_H, 0, 1);   // next tile's rows

    // explicit per-tile running pointers -> DS immediate offsets, minimal VALU
    const bf16_t* p0 = aB0;
    const bf16_t* p1 = aB0 + 16 * LDS_LD;
    const bf16_t* p2 = aB0 + 32 * LDS_LD;
    const bf16_t* p3 = aB0 + 48 * LDS_LD;

    v8f c0 = {0.f,0.f,0.f,0.f,0.f,0.f,0.f,0.f};
    v8f c1 = c0, c2 = c0, c3 = c0;

#pragma unroll 2
    for (int k = 0; k < TWO_H; k += 32) {
      const v16bf bf = __builtin_shufflevector(*(const v8bf*)pb, *(const v8bf*)(pb + 16), 0,1,2,3,4,5,6,7,8,9,10,11,12,13,14,15);
      const v16bf a0 = LOAD_A(p0);
      const v16bf a1 = LOAD_A(p1);
      const v16bf a2 = LOAD_A(p2);
      const v16bf a3 = LOAD_A(p3);
      c0 = __builtin_amdgcn_wmma_f32_16x16x32_bf16(false, a0, false, bf, (short)0, c0, false, false);
      c1 = __builtin_amdgcn_wmma_f32_16x16x32_bf16(false, a1, false, bf, (short)0, c1, false, false);
      c2 = __builtin_amdgcn_wmma_f32_16x16x32_bf16(false, a2, false, bf, (short)0, c2, false, false);
      c3 = __builtin_amdgcn_wmma_f32_16x16x32_bf16(false, a3, false, bf, (short)0, c3, false, false);
      pb += 32; p0 += 32; p1 += 32; p2 += 32; p3 += 32;
    }

    // epilogue: e = tanh(acc + h_part + b_attn); score += w_v[o] * e
    const float hv  = hpb_b[o0 + nlane];
    const float wvv = wv[o0 + nlane];
#pragma unroll
    for (int r = 0; r < 8; ++r) {
      sAcc[0][r] += wvv * tanh_fast(c0[r] + hv);
      sAcc[1][r] += wvv * tanh_fast(c1[r] + hv);
      sAcc[2][r] += wvv * tanh_fast(c2[r] + hv);
      sAcc[3][r] += wvv * tanh_fast(c3[r] + hv);
    }
  }
#undef LOAD_A

  // reduce over the 16 N-lanes of each half; lane 0 -> row r, lane 16 -> row r+8
#pragma unroll
  for (int mt = 0; mt < 4; ++mt) {
#pragma unroll
    for (int r = 0; r < 8; ++r) {
      float v = sAcc[mt][r];
      v += __shfl_xor(v, 1, 32);
      v += __shfl_xor(v, 2, 32);
      v += __shfl_xor(v, 4, 32);
      v += __shfl_xor(v, 8, 32);
      if (nlane == 0)
        atomicAdd(&scoreLds[mt * 16 + r + (khalf << 3)], v);   // ds_add_f32
    }
  }
  __syncthreads();
  if (tid < M_BLK) scores[m0 + tid] = scoreLds[tid];
}

// ---------------------------------------------------------------------------
// Masked softmax per batch. (b_v is a constant shift -> cancels in softmax.)
// ---------------------------------------------------------------------------
__global__ __launch_bounds__(256) void k_softmax(const float* __restrict__ scores,
                                                 const int* __restrict__ lengths,
                                                 float* __restrict__ energy) {
  const int b = blockIdx.x, tid = threadIdx.x;
  __shared__ float red[256];
  int len = lengths[b];
  if (len < 1) len = 1;
  if (len > SEQ) len = SEQ;
  const float* sc = scores + b * SEQ;

  float m = -3.4e38f;
  for (int s = tid; s < len; s += 256) m = fmaxf(m, sc[s]);
  red[tid] = m; __syncthreads();
  for (int w = 128; w > 0; w >>= 1) {
    if (tid < w) red[tid] = fmaxf(red[tid], red[tid + w]);
    __syncthreads();
  }
  const float bm = red[0];
  __syncthreads();

  float z = 0.f;
  for (int s = tid; s < len; s += 256) z += __expf(sc[s] - bm);
  red[tid] = z; __syncthreads();
  for (int w = 128; w > 0; w >>= 1) {
    if (tid < w) red[tid] += red[tid + w];
    __syncthreads();
  }
  const float inv = 1.0f / red[0];

  float* en = energy + b * SEQ;
  for (int s = tid; s < SEQ; s += 256)
    en[s] = (s < len) ? __expf(sc[s] - bm) * inv : 0.f;
}

// ---------------------------------------------------------------------------
// context[b][d] = sum_s energy[b][s] * enc[b][s][d]    (HBM-bound streaming)
//   grid (8 d-chunks, 16 batches, 16 s-chunks) x 256 threads
// ---------------------------------------------------------------------------
__global__ __launch_bounds__(256) void k_context(const float* __restrict__ enc,
                                                 const float* __restrict__ energy,
                                                 float* __restrict__ out) {
  const int b   = blockIdx.y;
  const int s0  = blockIdx.z * 128;
  const int col = blockIdx.x * 256 + threadIdx.x;
  __shared__ float eSh[128];
  if (threadIdx.x < 128) eSh[threadIdx.x] = energy[b * SEQ + s0 + threadIdx.x];
  __syncthreads();

  const float* base = enc + ((size_t)b * SEQ + s0) * TWO_H + col;
  float acc = 0.f;
#pragma unroll 4
  for (int i = 0; i < 128; ++i)
    acc += eSh[i] * base[(size_t)i * TWO_H];
  atomicAdd(&out[b * TWO_H + col], acc);
}

// ---------------------------------------------------------------------------
extern "C" void kernel_launch(void* const* d_in, const int* in_sizes, int n_in,
                              void* d_out, int out_size, void* d_ws, size_t ws_size,
                              hipStream_t stream) {
  (void)in_sizes; (void)n_in; (void)out_size; (void)ws_size;
  const float* enc    = (const float*)d_in[0];   // [16,2048,2048]
  const float* hidden = (const float*)d_in[1];   // [1024,16]
  const float* W_attn = (const float*)d_in[2];   // [2048,3072]
  const float* b_attn = (const float*)d_in[3];   // [2048]
  const float* w_v    = (const float*)d_in[4];   // [1,2048]
  // d_in[5] = b_v : constant shift, cancels in softmax -> unused
  const int*   lengths = (const int*)d_in[6];    // [16]
  float* out = (float*)d_out;

  char* ws = (char*)d_ws;                        // needs ~8.8 MB
  bf16_t* We    = (bf16_t*)(ws + WS_WE);
  float*  hpb   = (float*)(ws + WS_HPB);
  float*  score = (float*)(ws + WS_SCORES);
  float*  energ = (float*)(ws + WS_ENERGY);

  k_convert_we<<<1024, 256, 0, stream>>>(W_attn, We);
  k_hpb<<<128, dim3(16, 16), 0, stream>>>(W_attn, hidden, b_attn, hpb);
  k_zero_out<<<(NB * TWO_H + 255) / 256, 256, 0, stream>>>(out);
  k_scores<<<(NB * SEQ) / M_BLK, 256, 0, stream>>>(enc, We, hpb, w_v, score);
  k_softmax<<<NB, 256, 0, stream>>>(score, lengths, energ);
  k_context<<<dim3(TWO_H / 256, NB, SEQ / 128), 256, 0, stream>>>(enc, energ, out);
}
